// FeedBack_62234076119340
// MI455X (gfx1250) — compile-verified, matrix operand order checked by reading, not run on hardware
//
#include <hip/hip_runtime.h>
#include <hip/hip_bf16.h>

typedef __attribute__((ext_vector_type(16))) _Float16 v16h;
typedef __attribute__((ext_vector_type(8)))  _Float16 v8h;
typedef __attribute__((ext_vector_type(8)))  float    v8f;

// explicit global-address-space views (so laundered pointers still lower to
// global_load_b128, not flat_load_b128 which also ties up DScnt/LDS path)
typedef __attribute__((address_space(1))) const _Float16 GHalf;
typedef __attribute__((address_space(1))) const v8h     GV8h;

#define UNITS     256
#define BATCH     1024
#define T_IN      128
#define F_IN      3
#define OUT_STEPS 32
#define TILE_B    32
#define HS        264          // h row stride in halves (528B -> LDS bank step 4)
#define TOTAL_STEPS 159        // 128 warmup + 31 decode

// ---------------------------------------------------------------------------
// Prep: Wr f32 [256][1024] -> f16, pre-swizzled into WMMA-B fragment layout.
// Fragment (ntile 0..63, ktile 0..7) is 32 lanes x 16 halves, lane-contiguous,
// so the main kernel fetches a whole B fragment with two global_load_b128.
// Per CDNA5 ISA: lane l -> n = l%16 ; halves j: base=(l/16)*8,
//   j<8 : k = base+j ; j>=8 : k = 16+base+(j-8)   (within the 32-wide K tile)
// ---------------------------------------------------------------------------
__global__ void wr_swizzle_kernel(const float* __restrict__ Wr,
                                  _Float16* __restrict__ dst) {
  int tid = blockIdx.x * blockDim.x + threadIdx.x;   // 0 .. 262143
  if (tid >= UNITS * 4 * UNITS) return;
  int j    = tid & 15;
  int l    = (tid >> 4) & 31;
  int kt   = (tid >> 9) & 7;
  int nt   = tid >> 12;                               // 0..63
  int n    = nt * 16 + (l & 15);
  int base = (l >> 4) * 8;
  int kk   = (j < 8) ? (base + j) : (16 + base + (j - 8));
  int k    = kt * 32 + kk;
  dst[tid] = (_Float16)Wr[k * 1024 + n];
}

__device__ __forceinline__ float fast_sigmoid(float x) {
  return 1.0f / (1.0f + __expf(-x));
}
__device__ __forceinline__ float fast_tanh(float x) {
  // exact at saturation: exp(2x)->inf => 1 ; exp(2x)->0 => -1
  return 1.0f - 2.0f / (__expf(2.0f * x) + 1.0f);
}

__device__ __forceinline__ v16h cat8(v8h lo, v8h hi) {
  return __builtin_shufflevector(lo, hi, 0, 1, 2, 3, 4, 5, 6, 7,
                                 8, 9, 10, 11, 12, 13, 14, 15);
}

// ---------------------------------------------------------------------------
// Persistent LSTM kernel: one workgroup = 32 batch rows, 256 threads = 8 waves.
// Wave w owns units u in [32w, 32w+32). Pass ih in {0,1} handles the 16-unit
// sub-half (2w+ih): n-tiles { g*16 + 2w + ih } for gates g=0..3 -> i/f/g/o of
// a (row, unit) pair live in ONE lane; acc pressure is 64 VGPRs per pass.
// ---------------------------------------------------------------------------
__global__ void __launch_bounds__(256)
lstm_feedback_kernel(const float* __restrict__ inputs,   // [B][128][3]
                     const float* __restrict__ Wk,       // [3][1024]
                     const float* __restrict__ bias,     // [1024]
                     const float* __restrict__ Wd,       // [256][3]
                     const float* __restrict__ bd,       // [3]
                     const _Float16* __restrict__ wr_sw, // swizzled Wr, f16
                     float* __restrict__ out) {          // [B][32][3]
  __shared__ _Float16 h_lds[TILE_B * HS];
  __shared__ float    x_lds[TILE_B * 4];
  __shared__ float    wd_lds[UNITS * 3];
  __shared__ float    bd_lds[3];

  const int tid  = threadIdx.x;
  const int lane = tid & 31;
  const int w    = tid >> 5;        // wave 0..7
  const int hi   = lane >> 4;       // half-wave 0/1
  const int ln   = lane & 15;
  const int b0   = blockIdx.x * TILE_B;

  // decoder weights -> LDS (read every decode step)
  for (int i = tid; i < UNITS * 3; i += 256) wd_lds[i] = Wd[i];
  if (tid < 3) bd_lds[tid] = bd[tid];
  // h0 = 0
  for (int i = tid; i < TILE_B * HS; i += 256) h_lds[i] = (_Float16)0.0f;

  // per-lane gate-column constants: col(g,ih) = g*256 + (2w+ih)*16 + ln
  float bias_r[4][2];
  float wk_r[4][2][3];
#pragma unroll
  for (int g = 0; g < 4; ++g) {
#pragma unroll
    for (int i = 0; i < 2; ++i) {
      int col = g * 256 + (2 * w + i) * 16 + ln;
      bias_r[g][i] = bias[col];
#pragma unroll
      for (int f = 0; f < 3; ++f) wk_r[g][i][f] = Wk[f * 1024 + col];
    }
  }

  // cell state in registers: c_reg[mt][ih][r] <-> row m = mt*16 + r + 8*hi,
  // unit u = (2w+ih)*16 + ln  (matches WMMA C/D layout)
  float c_reg[2][2][8];
#pragma unroll
  for (int mt = 0; mt < 2; ++mt)
#pragma unroll
    for (int i = 0; i < 2; ++i)
#pragma unroll
      for (int r = 0; r < 8; ++r) c_reg[mt][i][r] = 0.0f;

  // Opaque integer copy of the Wr address: redefined by empty asm every step
  // so B loads are NOT loop-invariant (no LICM -> no 512-VGPR spill), then
  // cast to an addrspace(1) pointer so they stay global_load_b128.
  uintptr_t wr_base = (uintptr_t)wr_sw;

  for (int step = 0; step < TOTAL_STEPS; ++step) {
    asm volatile("" : "+s"(wr_base));   // zero-cost LICM blocker
    GHalf* wr_p = (GHalf*)wr_base;

    // warmup: feed inputs[:, step, :]; decode: x_lds was written by pred phase
    if (step < T_IN && tid < TILE_B * F_IN) {
      int m = tid / 3, f = tid - 3 * m;
      x_lds[m * 4 + f] =
          inputs[(size_t)(b0 + m) * (T_IN * F_IN) + step * F_IN + f];
    }
    __syncthreads();   // h(t-1) and x(t) visible

    float hnew[2][2][8];   // [mt][ih][r]

#pragma unroll
    for (int ih = 0; ih < 2; ++ih) {
      // ---- pass ih: z(:, units (2w+ih)*16+ln, all 4 gates) = h @ Wr ----
      v8f acc[2][4];       // [mt][gate]
#pragma unroll
      for (int mt = 0; mt < 2; ++mt)
#pragma unroll
        for (int g = 0; g < 4; ++g)
          acc[mt][g] = (v8f){0.f, 0.f, 0.f, 0.f, 0.f, 0.f, 0.f, 0.f};

#pragma unroll
      for (int kt = 0; kt < 8; ++kt) {
        v16h afrag[2];
#pragma unroll
        for (int mt = 0; mt < 2; ++mt) {
          const _Float16* ap = &h_lds[(mt * 16 + ln) * HS + kt * 32 + hi * 8];
          afrag[mt] = cat8(*(const v8h*)ap, *(const v8h*)(ap + 16));
        }
#pragma unroll
        for (int g = 0; g < 4; ++g) {
          const int gnt = g * 16 + 2 * w + ih;           // global n-tile
          GHalf* bp = wr_p + ((size_t)(gnt * 8 + kt) * 512) + lane * 16;
          v16h bfrag = cat8(((GV8h*)bp)[0], ((GV8h*)bp)[1]);
          acc[0][g] = __builtin_amdgcn_wmma_f32_16x16x32_f16(
              false, afrag[0], false, bfrag, (short)0, acc[0][g], false, false);
          acc[1][g] = __builtin_amdgcn_wmma_f32_16x16x32_f16(
              false, afrag[1], false, bfrag, (short)0, acc[1][g], false, false);
        }
      }

      // ---- gates for this pass, entirely in registers ----
#pragma unroll
      for (int mt = 0; mt < 2; ++mt) {
#pragma unroll
        for (int r = 0; r < 8; ++r) {
          const int m = mt * 16 + r + 8 * hi;
          const float x0 = x_lds[m * 4 + 0];
          const float x1 = x_lds[m * 4 + 1];
          const float x2 = x_lds[m * 4 + 2];
          float zi = acc[mt][0][r] + bias_r[0][ih] + x0 * wk_r[0][ih][0] +
                     x1 * wk_r[0][ih][1] + x2 * wk_r[0][ih][2];
          float zf = acc[mt][1][r] + bias_r[1][ih] + x0 * wk_r[1][ih][0] +
                     x1 * wk_r[1][ih][1] + x2 * wk_r[1][ih][2];
          float zg = acc[mt][2][r] + bias_r[2][ih] + x0 * wk_r[2][ih][0] +
                     x1 * wk_r[2][ih][1] + x2 * wk_r[2][ih][2];
          float zo = acc[mt][3][r] + bias_r[3][ih] + x0 * wk_r[3][ih][0] +
                     x1 * wk_r[3][ih][1] + x2 * wk_r[3][ih][2];
          float cn = fast_sigmoid(zf) * c_reg[mt][ih][r] +
                     fast_sigmoid(zi) * fast_tanh(zg);
          c_reg[mt][ih][r] = cn;
          hnew[mt][ih][r]  = fast_sigmoid(zo) * fast_tanh(cn);
        }
      }
    }

    __syncthreads();   // every wave finished reading h(t-1)
#pragma unroll
    for (int mt = 0; mt < 2; ++mt)
#pragma unroll
      for (int i = 0; i < 2; ++i)
#pragma unroll
        for (int r = 0; r < 8; ++r) {
          const int m = mt * 16 + r + 8 * hi;
          const int u = (2 * w + i) * 16 + ln;
          h_lds[m * HS + u] = (_Float16)hnew[mt][i][r];
        }
    __syncthreads();   // h(t) visible

    // ---- decode head: pred = h @ Wd + bd, all 256 threads + shfl reduce ----
    if (step >= T_IN - 1) {
      const int s  = step - (T_IN - 1);   // 0..31
      const int m  = tid >> 3;            // 0..31
      const int t8 = tid & 7;             // 8 threads per row
      float p0 = 0.f, p1 = 0.f, p2 = 0.f;
#pragma unroll 8
      for (int uu = 0; uu < 32; ++uu) {
        const int u  = t8 * 32 + uu;
        const float hv = (float)h_lds[m * HS + u];
        p0 += hv * wd_lds[u * 3 + 0];
        p1 += hv * wd_lds[u * 3 + 1];
        p2 += hv * wd_lds[u * 3 + 2];
      }
#pragma unroll
      for (int off = 4; off > 0; off >>= 1) {
        p0 += __shfl_down(p0, off, 8);
        p1 += __shfl_down(p1, off, 8);
        p2 += __shfl_down(p2, off, 8);
      }
      if (t8 == 0) {
        p0 += bd_lds[0];
        p1 += bd_lds[1];
        p2 += bd_lds[2];
        const size_t ob = (size_t)(b0 + m) * (OUT_STEPS * 3) + (size_t)s * 3;
        out[ob + 0] = p0;
        out[ob + 1] = p1;
        out[ob + 2] = p2;
        x_lds[m * 4 + 0] = p0;   // autoregressive feedback
        x_lds[m * 4 + 1] = p1;
        x_lds[m * 4 + 2] = p2;
      }
    }
    // loop-top __syncthreads() separates this phase from the next GEMM
  }
}

extern "C" void kernel_launch(void* const* d_in, const int* in_sizes, int n_in,
                              void* d_out, int out_size, void* d_ws,
                              size_t ws_size, hipStream_t stream) {
  (void)in_sizes; (void)n_in; (void)out_size; (void)ws_size;
  const float* inputs = (const float*)d_in[0];
  const float* Wk     = (const float*)d_in[1];
  const float* Wr     = (const float*)d_in[2];
  const float* b      = (const float*)d_in[3];
  const float* Wd     = (const float*)d_in[4];
  const float* bd     = (const float*)d_in[5];
  // d_in[6] = out_steps (device scalar) — fixed at 32, matching OUT_STEPS.
  float* out = (float*)d_out;

  _Float16* wr_sw = (_Float16*)d_ws;            // needs 512 KB of workspace

  // 1) swizzle Wr into f16 WMMA-B fragments (L2-resident thereafter)
  wr_swizzle_kernel<<<(UNITS * 4 * UNITS + 255) / 256, 256, 0, stream>>>(Wr,
                                                                         wr_sw);
  // 2) persistent recurrence: 32 workgroups × 256 threads (8 waves)
  lstm_feedback_kernel<<<BATCH / TILE_B, 256, 0, stream>>>(inputs, Wk, b, Wd,
                                                           bd, wr_sw, out);
}
